// SpeciesGNN_SoftForms_84834194030608
// MI455X (gfx1250) — matrix operand order repaired and checked by reading, not compiled, not run on hardware
//
#include <hip/hip_runtime.h>
#include <math.h>

typedef __attribute__((ext_vector_type(16))) _Float16 v16h;
typedef __attribute__((ext_vector_type(8)))  float    v8f;

// Problem constants: B=4, T=128, N=96 (NV=92,NH=4), D=32, H=32, TOP_K=8
#define NN 96
#define HH 32
#define TOPK 8
#define NPAIR (NN*NN)          // 9216
#define NBT (4*128)            // 512

// workspace layout (float offsets)
#define WS_ALPHA 0
#define WS_W     96                      // 5*9216 mixed form weights
#define WS_EMBI  (WS_W + 5*NPAIR)        // 3072 (emb_i@w1 + b1)
#define WS_EMBJ  (WS_EMBI + 3072)        // 3072
#define WS_EQ    (WS_EMBJ + 3072)        // 3072
#define WS_EK    (WS_EQ + 3072)          // 3072
#define WS_W2H   (WS_EK + 3072)          // 512 floats = 1024 halves (w2 in f16)

__device__ __forceinline__ float geluf(float x) {
  return 0.5f * x * (1.0f + erff(x * 0.70710678118654752440f));
}
__device__ __forceinline__ float wredmax(float v) {
  #pragma unroll
  for (int m = 16; m > 0; m >>= 1) v = fmaxf(v, __shfl_xor(v, m, 32));
  return v;
}
__device__ __forceinline__ float wredsum(float v) {
  #pragma unroll
  for (int m = 16; m > 0; m >>= 1) v += __shfl_xor(v, m, 32);
  return v;
}

__global__ void setup_k(const float* __restrict__ fc, const float* __restrict__ fg,
                        const float* __restrict__ har, const float* __restrict__ se,
                        const float* __restrict__ w1, const float* __restrict__ b1,
                        const float* __restrict__ qw, const float* __restrict__ qb,
                        const float* __restrict__ kw, const float* __restrict__ kb,
                        const float* __restrict__ w2, float* __restrict__ ws) {
  const int tid = threadIdx.x;
  for (int t = tid; t < NN; t += 256) {
    float x = har[t];
    float sp = (x > 20.f) ? x : log1pf(expf(x));   // softplus
    ws[WS_ALPHA + t] = sp + 0.01f;
  }
  for (int idx = tid; idx < 5*NPAIR; idx += 256) {
    float g = fg[idx];
    ws[WS_W + idx] = fc[idx] / (1.f + expf(-g));    // coef * sigmoid(gate)
  }
  for (int idx = tid; idx < NN*HH; idx += 256) {
    int i = idx >> 5, kk = idx & 31;
    float s_ei = b1[kk], s_ej = 0.f, s_eq = qb[kk], s_ek = kb[kk];
    for (int d = 0; d < 32; ++d) {
      float e = se[i*32 + d];
      s_ei = fmaf(e, w1[(2+d)*32 + kk], s_ei);
      s_ej = fmaf(e, w1[(34+d)*32 + kk], s_ej);
      s_eq = fmaf(e, qw[(1+d)*32 + kk], s_eq);
      s_ek = fmaf(e, kw[(1+d)*32 + kk], s_ek);
    }
    ws[WS_EMBI + idx] = s_ei;
    ws[WS_EMBJ + idx] = s_ej;
    ws[WS_EQ + idx]   = s_eq;
    ws[WS_EK + idx]   = s_ek;
  }
  _Float16* w2h = (_Float16*)(ws + WS_W2H);
  for (int idx = tid; idx < HH*HH; idx += 256) w2h[idx] = (_Float16)w2[idx];
}

__global__ void __launch_bounds__(256) main_k(
    const float* __restrict__ state, const float* __restrict__ qw,
    const float* __restrict__ kw, const float* __restrict__ w1,
    const float* __restrict__ b2g, const float* __restrict__ w3g,
    const float* __restrict__ b3g, const float* __restrict__ rg,
    const float* __restrict__ ws, float* __restrict__ out) {
  extern __shared__ float smem[];
  float* S   = smem;            // 9216  : scores -> attn (in place)
  float* U   = S + NPAIR;       // 9216  : q|k first, then reused as f4
  float* EI  = U + NPAIR;       // 3072
  float* EJ  = EI + 3072;       // 3072
  float* xs  = EJ + 3072;       // 96
  float* hol = xs + NN;         // 96
  float* wxi = hol + NN;        // 32
  float* wxj = wxi + 32;        // 32
  float* b2s = wxj + 32;        // 32
  float* w3s = b2s + 32;        // 32

  const int tid  = threadIdx.x;
  const int lane = tid & 31;
  const int wave = tid >> 5;
  const int bt   = blockIdx.x;
  const float* xg   = state + bt*NN;
  float* outLR = out + (size_t)bt*NN;
  float* outAT = out + (size_t)NBT*NN + (size_t)bt*NPAIR;

  // ---- stage 0: scalars / tables
  for (int t = tid; t < NN; t += 256) {
    float x = xg[t];
    xs[t] = x;
    hol[t] = x / (1.f + ws[WS_ALPHA + t] * x);
  }
  if (tid < 32) {
    wxi[tid] = w1[tid];          // w1 row 0
    wxj[tid] = w1[32 + tid];     // w1 row 1
    b2s[tid] = b2g[tid];
    w3s[tid] = w3g[tid];
  }
  for (int idx = tid; idx < NN*HH; idx += 256) {
    EI[idx] = ws[WS_EMBI + idx];
    EJ[idx] = ws[WS_EMBJ + idx];
  }
  __syncthreads();

  // ---- stage 1: q,k (rank-1 update of precomputed emb projections)
  for (int idx = tid; idx < NN*HH; idx += 256) {
    int i = idx >> 5, kk = idx & 31;
    U[idx]          = fmaf(xs[i], qw[kk], ws[WS_EQ + idx]);
    U[NN*HH + idx]  = fmaf(xs[i], kw[kk], ws[WS_EK + idx]);
  }
  __syncthreads();

  // ---- stage 2: scores = q k^T / sqrt(32)
  for (int p = tid; p < NPAIR; p += 256) {
    int i = p / NN, j = p - i*NN;
    const float* qi = U + i*32;
    const float* kj = U + NN*HH + j*32;
    float acc = 0.f;
    #pragma unroll
    for (int kk = 0; kk < 32; ++kk) acc = fmaf(qi[kk], kj[kk], acc);
    S[p] = acc * 0.17677669529663687f;
  }
  __syncthreads();

  // ---- stage 3: per-row top-8 threshold + softmax (one wave per row)
  for (int row = wave; row < NN; row += 8) {
    float a0 = S[row*NN + lane];
    float a1 = S[row*NN + lane + 32];
    float a2 = S[row*NN + lane + 64];
    int r0 = 0, r1 = 0, r2 = 0;
    float rowmax = 0.f, thr = 0.f;
    for (int it = 0; it < TOPK; ++it) {
      float c0 = r0 ? -3.4e38f : a0;
      float c1 = r1 ? -3.4e38f : a1;
      float c2 = r2 ? -3.4e38f : a2;
      float m = wredmax(fmaxf(c0, fmaxf(c1, c2)));
      if (it == 0) rowmax = m;
      thr = m;
      int slot = (!r0 && a0 == m) ? 0 : (!r1 && a1 == m) ? 1 : (!r2 && a2 == m) ? 2 : -1;
      unsigned long long bal = __ballot(slot >= 0);
      int first = __ffsll(bal) - 1;
      if (lane == first) {
        if (slot == 0) r0 = 1; else if (slot == 1) r1 = 1; else r2 = 1;
      }
    }
    float e0 = (a0 >= thr) ? expf(a0 - rowmax) : 0.f;
    float e1 = (a1 >= thr) ? expf(a1 - rowmax) : 0.f;
    float e2 = (a2 >= thr) ? expf(a2 - rowmax) : 0.f;
    float inv = 1.f / wredsum(e0 + e1 + e2);
    float t0 = e0*inv, t1 = e1*inv, t2 = e2*inv;
    S[row*NN + lane]      = t0;
    S[row*NN + lane + 32] = t1;
    S[row*NN + lane + 64] = t2;
    outAT[row*NN + lane]      = t0;
    outAT[row*NN + lane + 32] = t1;
    outAT[row*NN + lane + 64] = t2;
  }

  // ---- stage 4: f4 MLP via WMMA. Each 16-pair group has a fixed i (96 = 6*16).
  const _Float16* w2h = (const _Float16*)(ws + WS_W2H);
  const int col = lane & 15;     // A row index / D column index
  const int hs  = lane >> 4;     // half-wave select
  v16h blo, bhi;                 // w2 cols 0-15 and 16-31, B layout (32x16 f16)
  #pragma unroll
  for (int d = 0; d < 8; ++d) {
    int k0 = ((d < 4) ? 2*d : 16 + 2*(d-4)) + 8*hs;
    blo[2*d]   = w2h[ k0   *32 + col];
    blo[2*d+1] = w2h[(k0+1)*32 + col];
    bhi[2*d]   = w2h[ k0   *32 + col + 16];
    bhi[2*d+1] = w2h[(k0+1)*32 + col + 16];
  }
  const float b3v = b3g[0];

  for (int g = wave; g < NPAIR/16; g += 8) {     // 576 groups
    int i = g / 6;
    int j = (g - i*6)*16 + col;                  // pair row = col -> j
    float xi  = xs[i];
    float xjv = xs[j];
    v16h a;                                      // gelu(h) in f16, A layout 16x32
    #pragma unroll
    for (int d = 0; d < 8; ++d) {
      int kb = ((d < 4) ? 2*d : 16 + 2*(d-4)) + 8*hs;
      float h0 = fmaf(xi, wxi[kb],   fmaf(xjv, wxj[kb],   EI[i*32+kb]   + EJ[j*32+kb]));
      float h1 = fmaf(xi, wxi[kb+1], fmaf(xjv, wxj[kb+1], EI[i*32+kb+1] + EJ[j*32+kb+1]));
      a[2*d]   = (_Float16)geluf(h0);
      a[2*d+1] = (_Float16)geluf(h1);
    }
    v8f c = {0.f,0.f,0.f,0.f,0.f,0.f,0.f,0.f};
    v8f d0 = __builtin_amdgcn_wmma_f32_16x16x32_f16(false, a, false, blo, (short)0, c, false, false);
    v8f d1 = __builtin_amdgcn_wmma_f32_16x16x32_f16(false, a, false, bhi, (short)0, c, false, false);
    // epilogue: + b2, gelu, * w3, reduce over 32 hidden cols (16 lanes x 2 cols)
    float part[8];
    #pragma unroll
    for (int rr = 0; rr < 8; ++rr) {
      float u0 = d0[rr] + b2s[col];
      float u1 = d1[rr] + b2s[col + 16];
      part[rr] = geluf(u0)*w3s[col] + geluf(u1)*w3s[col + 16];
    }
    #pragma unroll
    for (int m = 1; m <= 8; m <<= 1) {
      #pragma unroll
      for (int rr = 0; rr < 8; ++rr) part[rr] += __shfl_xor(part[rr], m, 32);
    }
    if (col == 0) {                              // lanes 0 and 16
      int base = g*16 + hs*8;                    // rows 0-7 / 8-15
      #pragma unroll
      for (int rr = 0; rr < 8; ++rr) U[base + rr] = part[rr] + b3v;
    }
  }
  __syncthreads();

  // ---- stage 5: msgs = sum_f W_f * form_f ; agg = sum_j attn*msgs ; log_ratio
  const float* W0 = ws + WS_W;
  for (int i = wave; i < NN; i += 8) {
    float xi = xs[i];
    float acc = 0.f;
    #pragma unroll
    for (int s = 0; s < 3; ++s) {
      int j = lane + 32*s;
      int idx = i*NN + j;
      float xjv = xs[j], hj = hol[j];
      float msg = W0[idx]*xjv
                + W0[NPAIR + idx]*(xi*xjv)
                + W0[2*NPAIR + idx]*hj
                + W0[3*NPAIR + idx]*(xi*hj)
                + W0[4*NPAIR + idx]*U[idx];
      acc = fmaf(S[idx], msg, acc);
    }
    acc = wredsum(acc);
    if (lane == 0) outLR[i] = rg[i] + acc;
  }
}

extern "C" void kernel_launch(void* const* d_in, const int* in_sizes, int n_in,
                              void* d_out, int out_size, void* d_ws, size_t ws_size,
                              hipStream_t stream) {
  (void)in_sizes; (void)n_in; (void)out_size; (void)ws_size;
  const float* state = (const float*)d_in[0];
  const float* se    = (const float*)d_in[1];
  const float* qw    = (const float*)d_in[2];
  const float* qb    = (const float*)d_in[3];
  const float* kw    = (const float*)d_in[4];
  const float* kb    = (const float*)d_in[5];
  const float* fc    = (const float*)d_in[6];
  const float* fg    = (const float*)d_in[7];
  const float* har   = (const float*)d_in[8];
  const float* w1    = (const float*)d_in[9];
  const float* b1    = (const float*)d_in[10];
  const float* w2    = (const float*)d_in[11];
  const float* b2    = (const float*)d_in[12];
  const float* w3    = (const float*)d_in[13];
  const float* b3    = (const float*)d_in[14];
  const float* r     = (const float*)d_in[15];
  float* ws  = (float*)d_ws;
  float* out = (float*)d_out;

  setup_k<<<1, 256, 0, stream>>>(fc, fg, har, se, w1, b1, qw, qb, kw, kb, w2, ws);

  // dynamic LDS: 2*9216 + 2*3072 + 2*96 + 4*32 floats = 24896 floats = 99584 B
  const size_t lds_bytes = (size_t)(2*NPAIR + 2*NN*HH + 2*NN + 4*32) * sizeof(float);
  main_k<<<NBT, 256, lds_bytes, stream>>>(state, qw, kw, w1, b2, w3, b3, r, ws, out);
}
